// W8A8SparseCompressedLinear_21251498180869
// MI455X (gfx1250) — 8-bit matrix operand order repaired and checked by reading, compiled, not run on hardware
//
#include <hip/hip_runtime.h>
#include <hip/hip_fp16.h>
#include <stdint.h>

#define OUT_F 11008
#define IN_F  4096
#define TOK   4096
#define NNZ   (OUT_F * IN_F / 2)

#define USE_ASYNC_LDS 1

typedef __attribute__((ext_vector_type(8))) int v8i;

__global__ void quant_kernel(const _Float16* __restrict__ x,
                             const _Float16* __restrict__ scale,
                             const int* __restrict__ offset,
                             char* __restrict__ xq) {
    const size_t i = ((size_t)blockIdx.x * blockDim.x + threadIdx.x) * 8;
    const float s   = (float)scale[0];
    const float off = (float)offset[0];

    union { int4 v; _Float16 h[8]; } u;
    u.v = *(const int4*)(x + i);

    int q[8];
#pragma unroll
    for (int j = 0; j < 8; ++j) {
        float f = (float)u.h[j];
        float r = rintf(f / s) + off;
        r = fminf(fmaxf(r, -128.0f), 127.0f);
        q[j] = (int)r;
    }
    int lo = (q[0] & 255) | ((q[1] & 255) << 8) | ((q[2] & 255) << 16) | ((q[3] & 255) << 24);
    int hi = (q[4] & 255) | ((q[5] & 255) << 8) | ((q[6] & 255) << 16) | ((q[7] & 255) << 24);
    *(int2*)(xq + i) = make_int2(lo, hi);
}

__global__ void scatter_kernel(const int* __restrict__ vals,
                               const int* __restrict__ idx,
                               char* __restrict__ wdense) {
    const size_t i = (size_t)blockIdx.x * blockDim.x + threadIdx.x;
    wdense[(unsigned)idx[i]] = (char)vals[i];
}

__device__ __forceinline__ void async_b128_to_lds(unsigned lds_off, const void* gptr) {
    asm volatile("global_load_async_to_lds_b128 %0, %1, off"
                 :: "v"(lds_off), "v"((unsigned long long)(uintptr_t)gptr)
                 : "memory");
}
__device__ __forceinline__ void wait_asynccnt0() {
    asm volatile("s_wait_asynccnt 0x0" ::: "memory");
}

__global__ __launch_bounds__(256)
void gemm_kernel(const char* __restrict__ xq,
                 const char* __restrict__ wd,
                 const int* __restrict__ quant_bias,
                 const float* __restrict__ deq_scale,
                 _Float16* __restrict__ y) {
    __shared__ __align__(16) char smem[2][16384];

    const int tid   = threadIdx.x;
    const int lane  = tid & 31;
    const int wv    = tid >> 5;
    const int m_off = (wv & 3) * 32;
    const int n_off = (wv >> 2) * 64;
    const int lrow  = lane & 15;
    const int lhi   = lane >> 4;

    const int mBase = blockIdx.y * 128;
    const int nBase = blockIdx.x * 128;

    const int crow0 = tid >> 2;
    const int ccol  = (tid & 3) << 4;

    const char* gA = xq + (size_t)(mBase + crow0) * IN_F + ccol;
    const char* gB = wd + (size_t)(nBase + crow0) * IN_F + ccol;

#if USE_ASYNC_LDS
    const unsigned smemOff = (unsigned)(uintptr_t)&smem[0][0];
#endif

    const int NK = IN_F / 64;
    const v8i vzero = {0, 0, 0, 0, 0, 0, 0, 0};
    v8i acc[2][4];
#pragma unroll
    for (int mt = 0; mt < 2; ++mt)
#pragma unroll
        for (int nt = 0; nt < 4; ++nt)
            acc[mt][nt] = vzero;

#if USE_ASYNC_LDS
    auto issueTile = [&](int kt, int b) {
        const unsigned base = smemOff + (unsigned)b * 16384u;
        const size_t kB = (size_t)kt * 64;
        async_b128_to_lds(base + (unsigned)tid * 16u,           gA + kB);
        async_b128_to_lds(base + (unsigned)(tid + 256) * 16u,   gA + kB + (size_t)64 * IN_F);
        async_b128_to_lds(base + 8192u + (unsigned)tid * 16u,         gB + kB);
        async_b128_to_lds(base + 8192u + (unsigned)(tid + 256) * 16u, gB + kB + (size_t)64 * IN_F);
    };

    issueTile(0, 0);
#endif

#pragma unroll 1
    for (int kt = 0; kt < NK; ++kt) {
        const int b = kt & 1;

#if USE_ASYNC_LDS
        wait_asynccnt0();
        __syncthreads();
        if (kt + 1 < NK)
            issueTile(kt + 1, b ^ 1);
#else
        const size_t kB = (size_t)kt * 64;
        const int4 va0 = *(const int4*)(gA + kB);
        const int4 va1 = *(const int4*)(gA + kB + (size_t)64 * IN_F);
        const int4 vb0 = *(const int4*)(gB + kB);
        const int4 vb1 = *(const int4*)(gB + kB + (size_t)64 * IN_F);
        __syncthreads();
        ((int4*)&smem[b][0])[tid]          = va0;
        ((int4*)&smem[b][0])[tid + 256]    = va1;
        ((int4*)&smem[b][8192])[tid]       = vb0;
        ((int4*)&smem[b][8192])[tid + 256] = vb1;
        __syncthreads();
#endif

        const char* ldsA = &smem[b][0];
        const char* ldsB = &smem[b][8192];

        v8i bf[4];
#pragma unroll
        for (int nt = 0; nt < 4; ++nt) {
            const char* bp = ldsB + (size_t)(n_off + nt * 16 + lrow) * 64 + lhi * 16;
            const int4 b0 = *(const int4*)(bp);
            const int4 b1 = *(const int4*)(bp + 32);
            v8i bv;
            bv[0] = b0.x; bv[1] = b0.y; bv[2] = b0.z; bv[3] = b0.w;
            bv[4] = b1.x; bv[5] = b1.y; bv[6] = b1.z; bv[7] = b1.w;
            bf[nt] = bv;
        }

#pragma unroll
        for (int mt = 0; mt < 2; ++mt) {
            const char* ap = ldsA + (size_t)(m_off + mt * 16 + lrow) * 64 + lhi * 16;
            const int2 a0 = *(const int2*)(ap);
            const int2 a1 = *(const int2*)(ap + 8);
            const int2 a2 = *(const int2*)(ap + 32);
            const int2 a3 = *(const int2*)(ap + 40);
            v8i af;
            af[0] = a0.x; af[1] = a0.y; af[2] = a1.x; af[3] = a1.y;
            af[4] = a2.x; af[5] = a2.y; af[6] = a3.x; af[7] = a3.y;
#pragma unroll
            for (int nt = 0; nt < 4; ++nt) {
                acc[mt][nt] = __builtin_amdgcn_wmma_i32_16x16x64_iu8(
                     true, af,  true, bf[nt],
                    acc[mt][nt],  false,  false);
            }
        }
    }

#pragma unroll
    for (int nt = 0; nt < 4; ++nt) {
        const int n  = nBase + n_off + nt * 16 + lrow;
        const int qb = quant_bias[n];
        const float sc = deq_scale[n];
#pragma unroll
        for (int mt = 0; mt < 2; ++mt) {
            const int mrow = mBase + m_off + mt * 16 + lhi * 8;
#pragma unroll
            for (int r = 0; r < 8; ++r) {
                const float v = (float)(acc[mt][nt][r] + qb) * sc;
                y[(size_t)(mrow + r) * OUT_F + n] = (_Float16)v;
            }
        }
    }
}

extern "C" void kernel_launch(void* const* d_in, const int* in_sizes, int n_in,
                              void* d_out, int out_size, void* d_ws, size_t ws_size,
                              hipStream_t stream) {
    const _Float16* x            = (const _Float16*)d_in[0];
    const int*      weight_vals  = (const int*)d_in[1];
    const int*      index        = (const int*)d_in[2];
    const _Float16* input_scale  = (const _Float16*)d_in[3];
    const int*      input_offset = (const int*)d_in[4];
    const float*    deq_scale    = (const float*)d_in[5];
    const int*      quant_bias   = (const int*)d_in[6];
    _Float16*       y            = (_Float16*)d_out;

    char* xq = (char*)d_ws;
    char* wd = (char*)d_ws + (size_t)TOK * IN_F;

    hipMemsetAsync(wd, 0, (size_t)OUT_F * IN_F, stream);

    quant_kernel<<<(TOK * IN_F) / (256 * 8), 256, 0, stream>>>(x, input_scale, input_offset, xq);
    scatter_kernel<<<NNZ / 256, 256, 0, stream>>>(weight_vals, index, wd);

    dim3 grid(OUT_F / 128, TOK / 128);
    gemm_kernel<<<grid, 256, 0, stream>>>(xq, wd, quant_bias, deq_scale, y);
}
